// GATLayer_30279519437043
// MI455X (gfx1250) — matrix operand order repaired and checked by reading, not compiled, not run on hardware
//
#include <hip/hip_runtime.h>
#include <hip/hip_bf16.h>

#define ALPHA 0.2f
#define NEG_INF -10000000000.0f

typedef __attribute__((ext_vector_type(2)))  float  v2f;
typedef __attribute__((ext_vector_type(8)))  float  v8f;
typedef __attribute__((ext_vector_type(16))) __bf16 v16bf;

// Low 32 bits of a generic pointer to __shared__ == LDS byte address (ISA §10.2).
__device__ __forceinline__ unsigned lds_addr_of(const void* p) {
    return (unsigned)(unsigned long long)p;
}

// ---------------------------------------------------------------------------
// K1: Wh[h][n][u] = features[n][:] @ W[h][:][u]  (fp32 WMMA 16x16x4)
// grid = (N/16, H), block = 32 (one wave). Each block: 16 rows x 64 cols.
// ---------------------------------------------------------------------------
__global__ __launch_bounds__(32) void k1_wh(const float* __restrict__ F,
                                            const float* __restrict__ W,
                                            float* __restrict__ Wh) {
    const int h  = blockIdx.y;
    const int n0 = blockIdx.x * 16;
    const int l  = threadIdx.x;
    const int r  = l & 15;
    const int g  = l >> 4;

    v8f acc[4] = {};
    const float* Wp = W + (size_t)h * 128 * 64;

    for (int k0 = 0; k0 < 128; k0 += 4) {
        v2f a;
        const float* fp = F + (size_t)(n0 + r) * 128 + k0 + 2 * g;
        a[0] = fp[0];
        a[1] = fp[1];
#pragma unroll
        for (int c = 0; c < 4; ++c) {
            v2f b;
            b[0] = Wp[(size_t)(k0 + 2 * g) * 64 + 16 * c + r];
            b[1] = Wp[(size_t)(k0 + 2 * g + 1) * 64 + 16 * c + r];
            acc[c] = __builtin_amdgcn_wmma_f32_16x16x4_f32(
                false, a, false, b, (short)0, acc[c], false, false);
        }
    }

    float* out = Wh + ((size_t)h * 4096 + n0) * 64;
#pragma unroll
    for (int c = 0; c < 4; ++c)
#pragma unroll
        for (int v = 0; v < 8; ++v)
            out[(size_t)(v + 8 * g) * 64 + 16 * c + r] = acc[c][v];
}

// ---------------------------------------------------------------------------
// K1b: src/dst attention logit halves + bf16 transpose WhT[h][u][m].
// ---------------------------------------------------------------------------
__global__ void k1b(const float* __restrict__ Wh, const float* __restrict__ a,
                    float* __restrict__ src, float* __restrict__ dst,
                    __bf16* __restrict__ WhT) {
    const int id = blockIdx.x * blockDim.x + threadIdx.x; // 0..16383
    const int h = id >> 12;
    const int n = id & 4095;
    const float* wp = Wh + ((size_t)h * 4096 + n) * 64;
    const float* as = a + (size_t)h * 128;
    const float* ad = as + 64;
    float s = 0.f, d = 0.f;
#pragma unroll 8
    for (int u = 0; u < 64; ++u) {
        float w = wp[u];
        s += w * as[u];
        d += w * ad[u];
        WhT[(size_t)(h * 64 + u) * 4096 + n] = (__bf16)w; // coalesced across threads per u
    }
    src[id] = s;
    dst[id] = d;
}

// ---------------------------------------------------------------------------
// K2: rowmax[h][n] = max_m ( adj[n][m]>0 ? lrelu(src+dst) : NEG_INF )
// one block (256 threads) per row n; int4 loads + prefetch on the 64MB stream.
// ---------------------------------------------------------------------------
__global__ __launch_bounds__(256) void k2_rowmax(const int* __restrict__ adj,
                                                 const float* __restrict__ src,
                                                 const float* __restrict__ dst,
                                                 float* __restrict__ rowmax) {
    const int n = blockIdx.x;
    const int tid = threadIdx.x;
    __shared__ float red[256];

    float s[4], mx[4];
#pragma unroll
    for (int h = 0; h < 4; ++h) {
        s[h]  = src[h * 4096 + n];
        mx[h] = NEG_INF;
    }
    const int* row = adj + (size_t)n * 4096;
    for (int m4 = tid * 4; m4 < 4096; m4 += 1024) {
        if (m4 + 1024 < 4096) __builtin_prefetch(row + m4 + 1024, 0, 0);
        const int4 av = *(const int4*)(row + m4);
        const int cc[4] = {av.x, av.y, av.z, av.w};
#pragma unroll
        for (int j = 0; j < 4; ++j) {
            if (cc[j] > 0) {
#pragma unroll
                for (int h = 0; h < 4; ++h) {
                    float e = s[h] + dst[h * 4096 + m4 + j];
                    e = e > 0.f ? e : ALPHA * e;
                    mx[h] = fmaxf(mx[h], e);
                }
            }
        }
    }
#pragma unroll
    for (int h = 0; h < 4; ++h) {
        red[tid] = mx[h];
        __syncthreads();
        for (int st = 128; st > 0; st >>= 1) {
            if (tid < st) red[tid] = fmaxf(red[tid], red[tid + st]);
            __syncthreads();
        }
        if (tid == 0) rowmax[h * 4096 + n] = red[0];
        __syncthreads();
    }
}

// ---------------------------------------------------------------------------
// K3: out = mean_h softmax-rows(P_h) @ Wh_h, via bf16 WMMA 16x16x32.
// grid = N/16 blocks; block = 128 threads = 4 waves, wave w handles head w.
// Adjacency (16x32 i32) + dst (4x32 f32) tiles double-buffered in LDS via
// GLOBAL_LOAD_ASYNC_TO_LDS_B128/B32 (ASYNCcnt), overlapping copy with WMMA.
// ---------------------------------------------------------------------------
__global__ __launch_bounds__(128) void k3_attn(const int* __restrict__ adj,
                                               const float* __restrict__ src,
                                               const float* __restrict__ dst,
                                               const float* __restrict__ rowmax,
                                               const __bf16* __restrict__ WhT,
                                               float* __restrict__ out) {
    const int n0  = blockIdx.x * 16;
    const int tid = threadIdx.x;
    const int h   = tid >> 5;   // wave id == head
    const int l   = tid & 31;
    const int r   = l & 15;
    const int g   = l >> 4;

    __shared__ __align__(16) int   s_adj[2][16 * 32];
    __shared__ __align__(16) float s_dst[2][4 * 32];
    __shared__ float s_out[16 * 64];
    __shared__ float s_sum[4][16];

    const float srcv = src[h * 4096 + n0 + r];
    const float rm   = rowmax[h * 4096 + n0 + r];

    v8f acc[4] = {};
    float rowsum = 0.f;
    const __bf16* Bbase = WhT + (size_t)h * 64 * 4096;

    // Per-thread source/dest addresses for the async tile copies.
    const int idx = tid * 4;          // 4 ints of the 16x32 adjacency tile
    const int rr  = idx >> 5;
    const int cc  = idx & 31;
    const int*   gadj = adj + (size_t)(n0 + rr) * 4096 + cc;     // + m0 at issue
    const float* gdst = dst + (tid >> 5) * 4096 + (tid & 31);    // + m0 at issue

#define ISSUE_TILE(MT, BUF)                                                    \
    do {                                                                       \
        const int _m0 = (MT) * 32;                                             \
        const int*   _ga = gadj + _m0;                                         \
        const float* _gd = gdst + _m0;                                         \
        unsigned _la = lds_addr_of(&s_adj[(BUF)][idx]);                        \
        unsigned _ld = lds_addr_of(&s_dst[(BUF)][tid]);                        \
        asm volatile("global_load_async_to_lds_b128 %0, %1, off"               \
                     :: "v"(_la), "v"(_ga) : "memory");                        \
        asm volatile("global_load_async_to_lds_b32 %0, %1, off"                \
                     :: "v"(_ld), "v"(_gd) : "memory");                        \
    } while (0)

    ISSUE_TILE(0, 0);
    asm volatile("s_wait_asynccnt 0x0" ::: "memory");
    __syncthreads();

    for (int mt = 0; mt < 128; ++mt) {
        const int m0  = mt * 32;
        const int buf = mt & 1;

        if (mt + 1 < 128) ISSUE_TILE(mt + 1, buf ^ 1);  // prefetch next tile

        // Build P fragment in fp32, demote to bf16; masked entries underflow to 0.
        v16bf afrag;
        float lsum = 0.f;
#pragma unroll
        for (int e = 0; e < 16; ++e) {
            const int k = (e < 8) ? (8 * g + e) : (16 + 8 * g + (e - 8));
            float x = srcv + s_dst[buf][h * 32 + k];
            x = x > 0.f ? x : ALPHA * x;
            const float ev = (s_adj[buf][r * 32 + k] > 0) ? x : NEG_INF;
            const float p  = __expf(ev - rm);
            lsum += p;
            afrag[e] = (__bf16)p;
        }
        rowsum += lsum + __shfl_xor(lsum, 16, 32); // pair-reduce row halves

#pragma unroll
        for (int c = 0; c < 4; ++c) {
            union { v16bf v; unsigned u[8]; } b;
            const unsigned* q = (const unsigned*)(
                Bbase + (size_t)(16 * c + r) * 4096 + m0 + 8 * g);
#pragma unroll
            for (int v = 0; v < 8; ++v) b.u[v] = q[v < 4 ? v : v + 4];
            acc[c] = __builtin_amdgcn_wmma_f32_16x16x32_bf16(
                false, afrag, false, b.v, (short)0, acc[c], false, false);
        }

        // Next tile landed + everyone done with this buffer before reuse.
        asm volatile("s_wait_asynccnt 0x0" ::: "memory");
        __syncthreads();
    }
#undef ISSUE_TILE

    // Epilogue: normalize rows, reduce heads through LDS, write mean.
    if (l < 16) s_sum[h][l] = rowsum;
    for (int i = tid; i < 16 * 64; i += 128) s_out[i] = 0.f;
    __syncthreads();

#pragma unroll
    for (int c = 0; c < 4; ++c)
#pragma unroll
        for (int v = 0; v < 8; ++v) {
            const int row = v + 8 * g;
            const int col = 16 * c + r;
            atomicAdd(&s_out[row * 64 + col], acc[c][v] / s_sum[h][row]);
        }
    __syncthreads();

    for (int i = tid; i < 16 * 64; i += 128) {
        const int row = i >> 6, col = i & 63;
        out[(size_t)(n0 + row) * 64 + col] = s_out[i] * 0.25f;
    }
}

// ---------------------------------------------------------------------------
extern "C" void kernel_launch(void* const* d_in, const int* in_sizes, int n_in,
                              void* d_out, int out_size, void* d_ws, size_t ws_size,
                              hipStream_t stream) {
    const float* features  = (const float*)d_in[0];
    const int*   adjacency = (const int*)  d_in[1];
    const float* W         = (const float*)d_in[2];
    const float* a         = (const float*)d_in[3];

    char* ws = (char*)d_ws;
    float*  Wh     = (float*) (ws);                      // 4 MB  fp32 [h][n][u]
    __bf16* WhT    = (__bf16*)(ws + 4194304);            // 2 MB  bf16 [h][u][m]
    float*  src    = (float*) (ws + 6291456);            // 64 KB
    float*  dst    = (float*) (ws + 6356992);            // 64 KB
    float*  rowmax = (float*) (ws + 6422528);            // 64 KB
    float*  out    = (float*)d_out;                      // 4096 x 64 fp32

    k1_wh    <<<dim3(256, 4),   32, 0, stream>>>(features, W, Wh);
    k1b      <<<64,            256, 0, stream>>>(Wh, a, src, dst, WhT);
    k2_rowmax<<<4096,          256, 0, stream>>>(adjacency, src, dst, rowmax);
    k3_attn  <<<256,           128, 0, stream>>>(adjacency, src, dst, rowmax, WhT, out);
}
